// PatchEmbedding_SAPS_19877108646373
// MI455X (gfx1250) — matrix operand order repaired and checked by reading, not compiled, not run on hardware
//
#include <hip/hip_runtime.h>
#include <hip/hip_bf16.h>

typedef __attribute__((ext_vector_type(16))) _Float16 v16h;
typedef __attribute__((ext_vector_type(8)))  float    v8f;

#define NB     32      // batch
#define NC     3       // channels
#define HW     128     // H == W
#define NE     768     // embedding dim
#define NP     1024    // max_numpatches = (128/4)^2
#define NNODES 1365    // 1+4+16+64+256+1024
#define NLVL   6

__device__ __constant__ static const int OFF_[NLVL]  = {0, 1, 5, 21, 85, 341};
__device__ __constant__ static const int GTAB_[NLVL] = {1, 2, 4, 8, 16, 32};

static __device__ __forceinline__ int morton5(int r, int c) {
    int code = 0;
#pragma unroll
    for (int b = 0; b < 5; ++b)
        code |= (((r >> b) & 1) << (2 * b + 1)) | (((c >> b) & 1) << (2 * b));
    return code;
}

// ---------------------------------------------------------------------------
// Zero-fill the whole output buffer (tail slots must be 0; d_out is poisoned).
// ---------------------------------------------------------------------------
__global__ void saps_zero_kernel(float* __restrict__ p, size_t n) {
    size_t i = (size_t)blockIdx.x * blockDim.x + threadIdx.x;
    size_t stride = (size_t)gridDim.x * blockDim.x;
    for (; i < n; i += stride) p[i] = 0.0f;
}

// ---------------------------------------------------------------------------
// Quadtree control kernel: one 1024-thread block per batch.
//  - 4x4 cell sums of im_edge in LDS
//  - per-level node sums, split/leaf decisions (top-down)
//  - Morton bitmap of leaf top-left min-cells + LDS prefix scan == argsort
//  - writes nodeSlot[] (workspace) and dst/mask rows of d_out
// ---------------------------------------------------------------------------
__global__ void saps_quadtree_kernel(const float* __restrict__ im_edge,
                                     const int*   __restrict__ div_thres_p,
                                     int*   __restrict__ nodeSlot,
                                     float* __restrict__ outDst,
                                     float* __restrict__ outMask) {
    const int b = blockIdx.x;
    const int t = threadIdx.x;

    __shared__ float s4[NP];                 // 4x4 cell sums (32x32)
    __shared__ float nodeSum[NNODES];
    __shared__ unsigned char splitF[NNODES];
    __shared__ int   codeArr[NNODES];
    __shared__ int   bm[NP];                 // morton bitmap -> inclusive scan
    __shared__ int   own[NP];

    const float thres = (float)(*div_thres_p);

    // 1) 4x4 cell sums
    {
        const int ci = t >> 5, cj = t & 31;
        const float* eb = im_edge + (size_t)b * HW * HW;
        float acc = 0.0f;
#pragma unroll
        for (int y = 0; y < 4; ++y)
#pragma unroll
            for (int x = 0; x < 4; ++x)
                acc += eb[(ci * 4 + y) * HW + (cj * 4 + x)];
        s4[t] = acc;
        bm[t] = 0;
    }
    __syncthreads();

    // 2) per-level node sums from s4
    for (int l = 0; l < NLVL; ++l) {
        const int g = GTAB_[l];
        if (t < g * g) {
            const int cpn = 32 / g;
            const int i = t / g, j = t % g;
            float sum = 0.0f;
            for (int y = 0; y < cpn; ++y)
                for (int x = 0; x < cpn; ++x)
                    sum += s4[(i * cpn + y) * 32 + (j * cpn + x)];
            nodeSum[OFF_[l] + t] = sum;
        }
    }
    __syncthreads();

    // 3) top-down traversal: active -> split/leaf, leaf morton codes
    for (int l = 0; l < NLVL; ++l) {
        const int g = GTAB_[l];
        const int s = HW / g;
        if (t < g * g) {
            const int i = t / g, j = t % g;
            bool active = (l == 0)
                ? true
                : (splitF[OFF_[l - 1] + (i >> 1) * (g >> 1) + (j >> 1)] != 0);
            const bool canSplit = (s >= 8);            // (s/2) >= min_patchsize(4)
            const bool sp   = active && canSplit && (nodeSum[OFF_[l] + t] > thres);
            const bool leaf = active && !sp;
            splitF[OFF_[l] + t] = sp ? 1 : 0;
            int cd = -1;
            if (leaf) {
                cd = morton5(i * (s / 4), j * (s / 4));
                bm[cd] = 1;
            }
            codeArr[OFF_[l] + t] = cd;
        }
        __syncthreads();
    }

    // 4) inclusive scan of bm (Hillis-Steele); exclusive rank = bm - own
    own[t] = bm[t];
    __syncthreads();
    for (int d = 1; d < NP; d <<= 1) {
        const int v = (t >= d) ? bm[t - d] : 0;
        __syncthreads();
        bm[t] += v;
        __syncthreads();
    }

    // 5) emit nodeSlot + dst/mask rows (Z order == morton rank)
    for (int n = t; n < NNODES; n += NP) {
        const int cd = codeArr[n];
        int slot = -1;
        if (cd >= 0) {
            slot = bm[cd] - own[cd];
            int l = NLVL - 1;
#pragma unroll
            for (int q = 0; q < NLVL; ++q)
                if (n >= OFF_[q] && n < OFF_[q] + GTAB_[q] * GTAB_[q]) l = q;
            const int g = GTAB_[l], s = HW / g;
            const int idx = n - OFF_[l];
            const int i = idx / g, j = idx % g;
            float* dr = outDst + ((size_t)b * NP + slot) * 5;
            dr[0] = (float)HW; dr[1] = (float)HW; dr[2] = (float)s;
            dr[3] = (float)(i * s); dr[4] = (float)(j * s);
            outMask[(size_t)b * NP + slot] = 1.0f;
        }
        nodeSlot[b * NNODES + n] = slot;
    }
}

// ---------------------------------------------------------------------------
// Per-level patch-embedding GEMM using V_WMMA_F32_16X16X32_F16.
// Templated on patch size S: all index math strength-reduces to shifts; the
// K-tail guard exists only for S==4. Fragment loads are float4-vectorized:
//   A: two contiguous 8-float K-runs per lane (16B aligned for every S)
//   B: 16 contiguous floats of the weight row (+ prefetch of next K-slice)
// One wave computes one 16x16 output tile; block = (32,4) = 4 N-tiles.
// Tiles whose 16 rows contain no leaf are skipped wave-uniformly, so EXEC is
// all-ones at every WMMA (ISA 7.12 requirement). Store epilogue broadcasts
// the already-loaded slot ids via __shfl (ds_bpermute) instead of reloading.
// ---------------------------------------------------------------------------
template<int S>
__global__ void saps_wmma_gemm_kernel(const float* __restrict__ im,
                                      const float* __restrict__ w,
                                      const float* __restrict__ bias,
                                      const int*   __restrict__ nodeSlot,
                                      float* __restrict__ outEmb,
                                      int levelOff) {
    constexpr int G  = HW / S;
    constexpr int GG = G * G;
    constexpr int S2 = S * S;
    constexpr int K  = 3 * S2;

    const int lane  = threadIdx.x;               // 0..31 (wave32)
    const int tileM = blockIdx.x;
    const int tileN = blockIdx.y * 4 + threadIdx.y;

    // leaf slots for the 16 rows of this M-tile (lane r<16 -> row r);
    // wave-uniform early exit keeps EXEC all-ones at the WMMAs.
    int rowSlot;
    {
        const int r  = lane & 15;
        const int mm = tileM * 16 + r;
        const int bb = mm / GG, nn = mm % GG;
        rowSlot = nodeSlot[bb * NNODES + levelOff + nn];
        if (__ballot(rowSlot >= 0) == 0ull) return;
    }

    const int row  = lane & 15;                  // A-row / D-column lane id
    const int half = lane >> 4;
    const int m    = tileM * 16 + row;
    const int b    = m / GG;
    const int node = m % GG;
    const int pi   = (node / G) * S;             // patch top-left
    const int pj   = (node % G) * S;
    const float* imb = im + (size_t)b * NC * HW * HW;

    const int ncol = tileN * 16 + (lane & 15);   // global embedding column
    const float* wrow = w + (size_t)ncol * K;    // W[e][c][y][x] flattened

    v8f acc = (v8f){0.f, 0.f, 0.f, 0.f, 0.f, 0.f, 0.f, 0.f};

    for (int kb = 0; kb < K; kb += 32) {
        v16h afrag, bfrag;

        // ---- A fragment: two contiguous 8-float runs per lane ----
        // lane (l<16):  K in {kb+0..7,  kb+8..15}
        // lane (l>=16): K in {kb+16..23, kb+24..31}
#pragma unroll
        for (int grp = 0; grp < 2; ++grp) {
            const int k0 = kb + half * 16 + grp * 8;   // run start (mult of 8)
            float av[8];
            bool inb = true;
            if constexpr ((K & 31) != 0) inb = (k0 < K);
            if (inb) {
                const int c   = k0 / S2;               // constexpr strength-red.
                const int rem = k0 - c * S2;
                const int y   = rem / S;
                const int x   = rem - y * S;           // mult of 8 (S>=8) or 0
                const float* p = imb + ((size_t)c * HW + (pi + y)) * HW + (pj + x);
                if constexpr (S >= 8) {
                    const float4 lo = *(const float4*)p;        // one image row
                    const float4 hi = *(const float4*)(p + 4);
                    av[0]=lo.x; av[1]=lo.y; av[2]=lo.z; av[3]=lo.w;
                    av[4]=hi.x; av[5]=hi.y; av[6]=hi.z; av[7]=hi.w;
                } else {                                        // S==4: 2 rows
                    const float4 lo = *(const float4*)p;
                    const float4 hi = *(const float4*)(p + HW);
                    av[0]=lo.x; av[1]=lo.y; av[2]=lo.z; av[3]=lo.w;
                    av[4]=hi.x; av[5]=hi.y; av[6]=hi.z; av[7]=hi.w;
                }
            } else {
#pragma unroll
                for (int e = 0; e < 8; ++e) av[e] = 0.0f;
            }
#pragma unroll
            for (int e = 0; e < 8; ++e) afrag[grp * 8 + e] = (_Float16)av[e];
        }

        // ---- B fragment: 16 contiguous floats of weight row ----
        // lanes 0-15 hold K=kb..kb+15, lanes 16-31 hold K=kb+16..kb+31
        {
            const int k0 = kb + half * 16;
            float bv[16];
            bool inb = true;
            if constexpr ((K & 31) != 0) inb = (k0 < K);       // K mult of 16
            if (inb) {
                const float4 q0 = *(const float4*)(wrow + k0);
                const float4 q1 = *(const float4*)(wrow + k0 + 4);
                const float4 q2 = *(const float4*)(wrow + k0 + 8);
                const float4 q3 = *(const float4*)(wrow + k0 + 12);
                bv[0]=q0.x;  bv[1]=q0.y;  bv[2]=q0.z;  bv[3]=q0.w;
                bv[4]=q1.x;  bv[5]=q1.y;  bv[6]=q1.z;  bv[7]=q1.w;
                bv[8]=q2.x;  bv[9]=q2.y;  bv[10]=q2.z; bv[11]=q2.w;
                bv[12]=q3.x; bv[13]=q3.y; bv[14]=q3.z; bv[15]=q3.w;
            } else {
#pragma unroll
                for (int e = 0; e < 16; ++e) bv[e] = 0.0f;
            }
#pragma unroll
            for (int e = 0; e < 16; ++e) bfrag[e] = (_Float16)bv[e];
        }

        // prefetch next K-slice of the streaming weight row (global_prefetch_b8)
        if (kb + 32 < K)
            __builtin_prefetch(wrow + kb + 32 + half * 16, 0, 0);

        acc = __builtin_amdgcn_wmma_f32_16x16x32_f16(
            /*neg_a=*/false, afrag, /*neg_b=*/false, bfrag,
            /*c_mod=*/(short)0, acc, /*reuse_a=*/false, /*reuse_b=*/false);
    }

    // D layout: lane l element r -> D[(l>>4)*8 + r][l&15].
    // Slot ids come from the header loads via lane broadcast (no reloads).
    const float bval = bias[ncol];
#pragma unroll
    for (int r = 0; r < 8; ++r) {
        const int mr   = half * 8 + r;               // tile-local row
        const int mm   = tileM * 16 + mr;
        const int bb   = mm / GG;
        const int slot = __shfl(rowSlot, mr, 32);    // ds_bpermute broadcast
        if (slot >= 0)
            outEmb[((size_t)bb * NP + slot) * NE + ncol] = acc[r] + bval;
    }
}

// ---------------------------------------------------------------------------
// Launcher
// d_in order (setup_inputs dict): im, im_edge, w128,b128, w64,b64, w32,b32,
//                                 w16,b16, w8,b8, w4,b4, div_thres, min_patchsize
// d_out: emb (32*1024*768) ++ dst (32*1024*5) ++ mask (32*1024), as float.
// ---------------------------------------------------------------------------
template<int S>
static void launch_level(const float* im, const float* w, const float* bias,
                         const int* nodeSlot, float* outEmb, int levelOff,
                         hipStream_t stream) {
    constexpr int G = HW / S;
    constexpr int M = NB * G * G;                  // multiple of 16
    dim3 grid(M / 16, (NE / 16) / 4);
    dim3 block(32, 4);
    saps_wmma_gemm_kernel<S><<<grid, block, 0, stream>>>(
        im, w, bias, nodeSlot, outEmb, levelOff);
}

extern "C" void kernel_launch(void* const* d_in, const int* in_sizes, int n_in,
                              void* d_out, int out_size, void* d_ws, size_t ws_size,
                              hipStream_t stream) {
    (void)in_sizes; (void)n_in; (void)ws_size; (void)out_size;

    const float* im        = (const float*)d_in[0];
    const float* im_edge   = (const float*)d_in[1];
    const int*   div_thres = (const int*)d_in[14];

    float* outEmb  = (float*)d_out;
    float* outDst  = outEmb + (size_t)NB * NP * NE;
    float* outMask = outDst + (size_t)NB * NP * 5;

    int* nodeSlot = (int*)d_ws;                    // NB*NNODES ints (~171 KB)

    const size_t totalOut = (size_t)NB * NP * NE + (size_t)NB * NP * 5 + (size_t)NB * NP;

    // 1) zero the whole output (tail slots must be exactly zero)
    saps_zero_kernel<<<4096, 256, 0, stream>>>((float*)d_out, totalOut);

    // 2) quadtree control: leaf flags, Z-order slots, dst/mask
    saps_quadtree_kernel<<<NB, NP, 0, stream>>>(im_edge, div_thres,
                                                nodeSlot, outDst, outMask);

    // 3) per-level WMMA GEMMs (s = 128, 64, 32, 16, 8, 4)
    launch_level<128>(im, (const float*)d_in[2],  (const float*)d_in[3],
                      nodeSlot, outEmb, 0,   stream);
    launch_level<64>( im, (const float*)d_in[4],  (const float*)d_in[5],
                      nodeSlot, outEmb, 1,   stream);
    launch_level<32>( im, (const float*)d_in[6],  (const float*)d_in[7],
                      nodeSlot, outEmb, 5,   stream);
    launch_level<16>( im, (const float*)d_in[8],  (const float*)d_in[9],
                      nodeSlot, outEmb, 21,  stream);
    launch_level<8>(  im, (const float*)d_in[10], (const float*)d_in[11],
                      nodeSlot, outEmb, 85,  stream);
    launch_level<4>(  im, (const float*)d_in[12], (const float*)d_in[13],
                      nodeSlot, outEmb, 341, stream);
}
